// WindowAttention_13572096655432
// MI455X (gfx1250) — matrix operand order repaired and checked by reading, not compiled, not run on hardware
//
#include <hip/hip_runtime.h>
#include <cstdint>

#define DIMC 128
#define HC   32

typedef __attribute__((ext_vector_type(16))) __bf16 bf16x16;
typedef __attribute__((ext_vector_type(8)))  float  f32x8;

__device__ __forceinline__ unsigned short f2bf(float f) {
  unsigned int u = __builtin_bit_cast(unsigned int, f);
  u += 0x7fffu + ((u >> 16) & 1u);   // round-to-nearest-even
  return (unsigned short)(u >> 16);
}
__device__ __forceinline__ __bf16 u2b(unsigned short u) {
  return __builtin_bit_cast(__bf16, u);
}

// A operand: 16x32 (MxK). lane%16 = M row; VGPR v holds K pair
// (lane/16)*16 + v*2.
__device__ __forceinline__ bf16x16 load_A(const unsigned short* base, int row, int col, int ld) {
  int l = threadIdx.x & 31;
  int r = l & 15, h = l >> 4;
  const unsigned short* p = base + (row + r) * ld + col;
  bf16x16 a;
#pragma unroll
  for (int v = 0; v < 8; ++v) {
    int k0 = h * 16 + v * 2;
    a[2 * v]     = u2b(p[k0]);
    a[2 * v + 1] = u2b(p[k0 + 1]);
  }
  return a;
}

// B operand: 32x16 (KxN). lane%16 = column N; lanes 0-15 hold K=0..15,
// lanes 16-31 hold K=16..31. Source stored row-major by N (B transposed),
// so each lane reads 16 contiguous bf16 (-> b128 loads).
__device__ __forceinline__ bf16x16 load_Bt(const unsigned short* base, int nbase, int kbase, int ld) {
  int l = threadIdx.x & 31;
  const unsigned short* p = base + (nbase + (l & 15)) * ld + kbase + ((l >> 4) << 4);
  bf16x16 b;
#pragma unroll
  for (int e = 0; e < 16; ++e) b[e] = u2b(p[e]);
  return b;
}

// B operand from a K-major (row = K) source: strided reads.
__device__ __forceinline__ bf16x16 load_Bc(const unsigned short* base, int kbase, int ncol, int ld) {
  int l = threadIdx.x & 31;
  int kb = kbase + ((l >> 4) << 4);
  const unsigned short* p = base + ncol + (l & 15);
  bf16x16 b;
#pragma unroll
  for (int e = 0; e < 16; ++e) b[e] = u2b(p[(kb + e) * ld]);
  return b;
}

__device__ __forceinline__ f32x8 wmma_bf16(bf16x16 a, bf16x16 b, f32x8 c) {
  return __builtin_amdgcn_wmma_f32_16x16x32_bf16(false, a, false, b, (short)0, c, false, false);
}

// token (window wi, local token t) -> flat token index in [B,D,H,W]
template <int BR>
__device__ __forceinline__ int tokflat(int wi, int t, int D, int H, int W) {
  if (BR == 0) {                      // xy: [B][D][H/7][W/7], tok = hh*7+ww
    int nH = H / 7, nW = W / 7;
    int per = D * nH * nW;
    int b = wi / per, r = wi % per;
    int d = r / (nH * nW); int r2 = r % (nH * nW);
    int hb = r2 / nW, wb = r2 % nW;
    int hh = t / 7, ww = t % 7;
    return ((b * D + d) * H + hb * 7 + hh) * W + wb * 7 + ww;
  } else if (BR == 1) {               // th: [B][D/8][H/7][W], tok = dd*7+hh
    int nD = D / 8, nH = H / 7;
    int per = nD * nH * W;
    int b = wi / per, r = wi % per;
    int db = r / (nH * W); int r2 = r % (nH * W);
    int hb = r2 / W, w = r2 % W;
    int dd = t / 7, hh = t % 7;
    return ((b * D + db * 8 + dd) * H + hb * 7 + hh) * W + w;
  } else {                            // tw: [B][D/8][H][W/7], tok = dd*7+ww
    int nD = D / 8, nW = W / 7;
    int per = nD * H * nW;
    int b = wi / per, r = wi % per;
    int db = r / (H * nW); int r2 = r % (H * nW);
    int h = r2 / nW, wb = r2 % nW;
    int dd = t / 7, ww = t % 7;
    return ((b * D + db * 8 + dd) * H + h) * W + wb * 7 + ww;
  }
}

// Fused window attention: one block = one window, one wave = one head.
template <int BR, int NTOK>
__global__ __launch_bounds__(128) void win_attn(
    const float* __restrict__ x,
    const unsigned short* __restrict__ qkvT,   // [384][128] bf16 (transposed)
    const float* __restrict__ qkv_b,           // [384]
    const unsigned short* __restrict__ projT,  // [128][128] bf16 (transposed)
    const float* __restrict__ proj_b,          // [128]
    float* __restrict__ out,
    int D, int H, int W) {
  __shared__ __align__(16) unsigned short r0[64 * 128];  // xw (phase A), then Pbuf 4x[16][64]
  __shared__ __align__(16) unsigned short r1[64 * 128];  // qs, later obuf
  __shared__ __align__(16) unsigned short ks[64 * 128];
  __shared__ __align__(16) unsigned short vs[64 * 128];

  const int tid  = threadIdx.x;
  const int wid  = tid >> 5;      // head id
  const int lane = tid & 31;
  const int wi   = blockIdx.x;
  const float scale = 0.17677669529663687f;  // 32^-0.5

  // ---- load window tokens -> r0 as bf16 (rows >= NTOK zeroed) ----
  {
    int r  = tid >> 1;
    int c0 = (tid & 1) * 64;
    unsigned int* dst = (unsigned int*)&r0[r * DIMC + c0];
    if (r < NTOK) {
      long gb = (long)tokflat<BR>(wi, r, D, H, W) * DIMC + c0;
      const float4* src = (const float4*)(x + gb);
#pragma unroll
      for (int c = 0; c < 16; ++c) {
        float4 f = src[c];
        dst[2 * c]     = (unsigned int)f2bf(f.x) | ((unsigned int)f2bf(f.y) << 16);
        dst[2 * c + 1] = (unsigned int)f2bf(f.z) | ((unsigned int)f2bf(f.w) << 16);
      }
    } else {
#pragma unroll
      for (int c = 0; c < 32; ++c) dst[c] = 0u;
    }
  }
  __syncthreads();

  // ---- phase A: QKV = x @ Wqkv + b (each wave computes its own head's cols) ----
#pragma unroll
  for (int s = 0; s < 3; ++s) {          // 0=q 1=k 2=v
#pragma unroll
    for (int sub = 0; sub < 2; ++sub) {  // two 16-wide tiles per head
      int gcol = s * DIMC + wid * HC + sub * 16;
      bf16x16 Bk[4];
#pragma unroll
      for (int kk = 0; kk < 4; ++kk) Bk[kk] = load_Bt(qkvT, gcol, kk * 32, DIMC);
#pragma unroll
      for (int i = 0; i < 4; ++i) {      // M tiles
        f32x8 acc = {};
#pragma unroll
        for (int kk = 0; kk < 4; ++kk) {
          bf16x16 a = load_A(r0, i * 16, kk * 32, DIMC);
          acc = wmma_bf16(a, Bk[kk], acc);
        }
        int ncol = gcol + (lane & 15);
        float bias = qkv_b[ncol];
        unsigned short* dstbuf = (s == 0) ? r1 : (s == 1 ? ks : vs);
        int hc = ncol & 127;
        int mb = i * 16 + ((lane >> 4) << 3);
#pragma unroll
        for (int j = 0; j < 8; ++j) {
          float v = acc[j] + bias;
          if (s == 0) v *= scale;
          dstbuf[(mb + j) * DIMC + hc] = f2bf(v);
        }
      }
    }
  }
  __syncthreads();

  // ---- stage q tiles into registers, then r0/r1 get reused ----
  bf16x16 qA[4];
#pragma unroll
  for (int i = 0; i < 4; ++i) qA[i] = load_A(r1, i * 16, wid * HC, DIMC);
  __syncthreads();   // r1 becomes obuf; r0 becomes Pbuf

  // ---- phase B: per-head attention, softmax fully in registers ----
  {
    unsigned short* Pu = r0 + wid * (16 * 64);   // [16][64] bf16 per head
    bf16x16 kb[4], vb[4];
#pragma unroll
    for (int j = 0; j < 4; ++j) kb[j] = load_Bt(ks, j * 16, wid * HC, DIMC);
#pragma unroll
    for (int jn = 0; jn < 2; ++jn)
#pragma unroll
      for (int kk = 0; kk < 2; ++kk)
        vb[jn * 2 + kk] = load_Bc(vs, kk * 32, wid * HC + jn * 16, DIMC);

#pragma unroll 1
    for (int i = 0; i < 4; ++i) {
      // S = q_i @ k^T : 4 key tiles in registers (D layout: lane=key col, vgpr=row)
      f32x8 sAcc[4];
#pragma unroll
      for (int j = 0; j < 4; ++j) {
        f32x8 z = {};
        sAcc[j] = wmma_bf16(qA[i], kb[j], z);
      }
      // mask key columns >= NTOK
#pragma unroll
      for (int j = 0; j < 4; ++j) {
        bool valid = (j * 16 + (lane & 15)) < NTOK;
#pragma unroll
        for (int jj = 0; jj < 8; ++jj)
          sAcc[j][jj] = valid ? sAcc[j][jj] : -3.0e38f;
      }
      // row max across 4 tiles + 16 lanes (halves reduce their own rows)
      float mx[8], sum[8];
#pragma unroll
      for (int jj = 0; jj < 8; ++jj) {
        float m = fmaxf(fmaxf(sAcc[0][jj], sAcc[1][jj]), fmaxf(sAcc[2][jj], sAcc[3][jj]));
#pragma unroll
        for (int off = 1; off < 16; off <<= 1) m = fmaxf(m, __shfl_xor(m, off, 16));
        mx[jj] = m;
      }
      // exp + row sum
#pragma unroll
      for (int jj = 0; jj < 8; ++jj) {
        float s0 = __expf(sAcc[0][jj] - mx[jj]);
        float s1 = __expf(sAcc[1][jj] - mx[jj]);
        float s2 = __expf(sAcc[2][jj] - mx[jj]);
        float s3 = __expf(sAcc[3][jj] - mx[jj]);
        sAcc[0][jj] = s0; sAcc[1][jj] = s1; sAcc[2][jj] = s2; sAcc[3][jj] = s3;
        float su = (s0 + s1) + (s2 + s3);
#pragma unroll
        for (int off = 1; off < 16; off <<= 1) su += __shfl_xor(su, off, 16);
        sum[jj] = 1.0f / su;
      }
      // write normalized P (bf16) to LDS for the A-operand transpose
#pragma unroll
      for (int j = 0; j < 4; ++j) {
        int nc = j * 16 + (lane & 15);
        int mb = (lane >> 4) << 3;
#pragma unroll
        for (int jj = 0; jj < 8; ++jj)
          Pu[(mb + jj) * 64 + nc] = f2bf(sAcc[j][jj] * sum[jj]);
      }
      // O_i = P @ V
#pragma unroll
      for (int jn = 0; jn < 2; ++jn) {
        f32x8 oAcc = {};
#pragma unroll
        for (int kk = 0; kk < 2; ++kk) {
          bf16x16 pa = load_A(Pu, 0, kk * 32, 64);
          oAcc = wmma_bf16(pa, vb[jn * 2 + kk], oAcc);
        }
        int mb = i * 16 + ((lane >> 4) << 3);
        int nc = wid * HC + jn * 16 + (lane & 15);
#pragma unroll
        for (int jj = 0; jj < 8; ++jj) r1[(mb + jj) * DIMC + nc] = f2bf(oAcc[jj]);
      }
    }
  }
  __syncthreads();

  // ---- phase C: out = obuf @ Wproj + b; scatter store / atomic add ----
#pragma unroll
  for (int nt2 = 0; nt2 < 2; ++nt2) {
    int nt = wid * 2 + nt2;              // 8 N-tiles over 4 waves
    bf16x16 Bp[4];
#pragma unroll
    for (int kk = 0; kk < 4; ++kk) Bp[kk] = load_Bt(projT, nt * 16, kk * 32, DIMC);
#pragma unroll
    for (int i = 0; i < 4; ++i) {
      f32x8 acc = {};
#pragma unroll
      for (int kk = 0; kk < 4; ++kk) {
        bf16x16 a = load_A(r1, i * 16, kk * 32, DIMC);
        acc = wmma_bf16(a, Bp[kk], acc);
      }
      int gcol = nt * 16 + (lane & 15);
      float bias = proj_b[gcol];
      int mb = i * 16 + ((lane >> 4) << 3);
#pragma unroll
      for (int j = 0; j < 8; ++j) {
        int tok = mb + j;
        if (tok < NTOK) {
          long flat = (long)tokflat<BR>(wi, tok, D, H, W) * DIMC + gcol;
          float v = acc[j] + bias;
          if (BR == 0) out[flat] = v;                 // xy covers every element once
          else unsafeAtomicAdd(&out[flat], v);        // global_atomic_add_f32
        }
      }
    }
  }
}

// Transpose + f32->bf16 all six weight matrices into workspace (once per call).
__global__ void prep_weights(const float* qw0, const float* qw1, const float* qw2,
                             const float* pw0, const float* pw1, const float* pw2,
                             unsigned short* ws) {
  const int QN = 384 * 128, PN = 128 * 128;
  int idx = blockIdx.x * blockDim.x + threadIdx.x;
  if (idx < 3 * QN) {
    int m = idx / QN, r = idx % QN;
    int n = r / 128, k = r % 128;
    const float* w = (m == 0) ? qw0 : (m == 1 ? qw1 : qw2);
    ws[m * QN + r] = f2bf(w[k * 384 + n]);
  } else if (idx < 3 * QN + 3 * PN) {
    int i2 = idx - 3 * QN;
    int m = i2 / PN, r = i2 % PN;
    int n = r / 128, k = r % 128;
    const float* w = (m == 0) ? pw0 : (m == 1 ? pw1 : pw2);
    ws[3 * QN + m * PN + r] = f2bf(w[k * 128 + n]);
  }
}

extern "C" void kernel_launch(void* const* d_in, const int* in_sizes, int n_in,
                              void* d_out, int out_size, void* d_ws, size_t ws_size,
                              hipStream_t stream) {
  const float* x         = (const float*)d_in[0];
  const float* qkv_w     = (const float*)d_in[1];
  const float* qkv_b     = (const float*)d_in[2];
  const float* qkv_th_w  = (const float*)d_in[3];
  const float* qkv_th_b  = (const float*)d_in[4];
  const float* qkv_tw_w  = (const float*)d_in[5];
  const float* qkv_tw_b  = (const float*)d_in[6];
  const float* proj_w    = (const float*)d_in[7];
  const float* proj_b    = (const float*)d_in[8];
  const float* proj_th_w = (const float*)d_in[9];
  const float* proj_th_b = (const float*)d_in[10];
  const float* proj_tw_w = (const float*)d_in[11];
  const float* proj_tw_b = (const float*)d_in[12];
  float* out = (float*)d_out;

  const int D = 8, H = 56, W = 56;
  const int Bb = in_sizes[0] / (D * H * W * DIMC);

  unsigned short* ws = (unsigned short*)d_ws;
  const int QN = 384 * 128, PN = 128 * 128;
  int prepN = 3 * QN + 3 * PN;
  prep_weights<<<(prepN + 255) / 256, 256, 0, stream>>>(
      qkv_w, qkv_th_w, qkv_tw_w, proj_w, proj_th_w, proj_tw_w, ws);

  unsigned short* qT0 = ws;
  unsigned short* qT1 = ws + QN;
  unsigned short* qT2 = ws + 2 * QN;
  unsigned short* pT0 = ws + 3 * QN;
  unsigned short* pT1 = ws + 3 * QN + PN;
  unsigned short* pT2 = ws + 3 * QN + 2 * PN;

  int nw_xy = Bb * D * (H / 7) * (W / 7);         // 4096
  int nw_th = Bb * (D / 8) * (H / 7) * W;         // 3584
  int nw_tw = Bb * (D / 8) * H * (W / 7);         // 3584

  win_attn<0, 49><<<nw_xy, 128, 0, stream>>>(x, qT0, qkv_b,    pT0, proj_b,    out, D, H, W);
  win_attn<1, 56><<<nw_th, 128, 0, stream>>>(x, qT1, qkv_th_b, pT1, proj_th_b, out, D, H, W);
  win_attn<2, 56><<<nw_tw, 128, 0, stream>>>(x, qT2, qkv_tw_b, pT2, proj_tw_b, out, D, H, W);
}
